// GNNSAGE_88682484727895
// MI455X (gfx1250) — matrix operand order repaired and checked by reading, not compiled, not run on hardware
//
#include <hip/hip_runtime.h>
#include <hip/hip_bf16.h>

// ---------- WMMA types (CDNA5 / gfx1250, wave32) ----------
typedef __attribute__((ext_vector_type(16))) __bf16 bf16x16;
typedef __attribute__((ext_vector_type(2)))  __bf16 bf16x2;
typedef __attribute__((ext_vector_type(8)))  float  f32x8;

#define DFEAT 64

struct bf16x2x8 { bf16x2 p[8]; };

// packed float2 -> bf16x2: hardware v_cvt_pk_bf16_f32 (gfx12-lineage VOP3)
__device__ __forceinline__ bf16x2 cvt2(float a, float b) {
#if __has_builtin(__builtin_amdgcn_cvt_pk_bf16_f32)
    return __builtin_amdgcn_cvt_pk_bf16_f32(a, b);
#else
    unsigned int o;
    asm("v_cvt_pk_bf16_f32 %0, %1, %2" : "=v"(o) : "v"(a), "v"(b));
    return __builtin_bit_cast(bf16x2, o);
#endif
}

// ---------- kernel 1: zero workspace (agg + cnt are contiguous) ----------
__global__ void GNNSAGE_zero(float* __restrict__ p, long long n) {
    long long i = (long long)blockIdx.x * blockDim.x + threadIdx.x;
    long long stride = (long long)gridDim.x * blockDim.x;
    long long n4 = n >> 2;
    float4 z = {0.f, 0.f, 0.f, 0.f};
    for (long long k = i; k < n4; k += stride) ((float4*)p)[k] = z;
    for (long long k = (n4 << 2) + i; k < n; k += stride) p[k] = 0.0f;
}

// ---------- kernel 2: scatter-add of source features + degree count ----------
// thread t handles edge e = t/16, feature quad q = t%16 (4 contiguous floats).
// agg (25.6MB) fits in the 192MB L2 -> atomics resolve at L2 throughput.
__global__ void GNNSAGE_scatter(const float* __restrict__ feat,
                                const long long* __restrict__ ei, // [2,E]
                                float* __restrict__ agg,
                                float* __restrict__ cnt,
                                int E) {
    long long t = (long long)blockIdx.x * blockDim.x + threadIdx.x;
    int e = (int)(t >> 4);
    int q = (int)(t & 15);
    if (e >= E) return;
    long long s = ei[e];
    long long d = ei[(long long)E + e];
    const float4 v = ((const float4*)(feat + s * DFEAT))[q];
    float* a = agg + d * DFEAT + q * 4;
    atomicAdd(a + 0, v.x);
    atomicAdd(a + 1, v.y);
    atomicAdd(a + 2, v.z);
    atomicAdd(a + 3, v.w);
    if (q == 0) atomicAdd(cnt + d, 1.0f);
}

// ---------- A-operand loader: 16x32 bf16 tile for one K-half ----------
// lane (m = lane&15, hi = lane>>4): elems 0-15 -> K=k0+hi*16+e.
// One contiguous 16-float segment.
__device__ __forceinline__ bf16x16 load_a(const float* __restrict__ row,
                                          float scale, int k0, int hi) {
    const float4* p = (const float4*)row;
    int b0 = (k0 + hi * 16) >> 2;
    int b1 = b0 + 2;
    float4 s0 = p[b0], s1 = p[b0 + 1], s2 = p[b1], s3 = p[b1 + 1];
    bf16x2x8 u;
    u.p[0] = cvt2(s0.x * scale, s0.y * scale);
    u.p[1] = cvt2(s0.z * scale, s0.w * scale);
    u.p[2] = cvt2(s1.x * scale, s1.y * scale);
    u.p[3] = cvt2(s1.z * scale, s1.w * scale);
    u.p[4] = cvt2(s2.x * scale, s2.y * scale);
    u.p[5] = cvt2(s2.z * scale, s2.w * scale);
    u.p[6] = cvt2(s3.x * scale, s3.y * scale);
    u.p[7] = cvt2(s3.z * scale, s3.w * scale);
    return __builtin_bit_cast(bf16x16, u);
}

// ---------- kernel 3: fused SAGE linear: out = mean@Wl^T + b + x@Wr^T ----------
// 256 threads = 8 wave32; each wave owns a 16-node tile and issues 16
// v_wmma_f32_16x16x32_bf16 (4 j-tiles x 2 K-halves x 2 matrices).
// Weights are converted to bf16 in LDS once per block; B operands are then
// two ds_load_b128 each (B = W^T realized purely via addressing).
template <int RELU>
__global__ void __launch_bounds__(256)
GNNSAGE_linear(const float* __restrict__ xin,
               const float* __restrict__ agg,
               const float* __restrict__ cnt,
               const float* __restrict__ wl,
               const float* __restrict__ wr,
               const float* __restrict__ bias,
               float* __restrict__ out,
               int N) {
    __shared__ __align__(32) __bf16 swl[DFEAT * DFEAT];   // 8KB
    __shared__ __align__(32) __bf16 swr[DFEAT * DFEAT];   // 8KB

    // stage weights as bf16 into LDS (all threads participate, then barrier)
    for (int i = threadIdx.x; i < (DFEAT * DFEAT) / 2; i += 256) {
        float2 a = ((const float2*)wl)[i];
        ((bf16x2*)swl)[i] = cvt2(a.x, a.y);
        float2 c = ((const float2*)wr)[i];
        ((bf16x2*)swr)[i] = cvt2(c.x, c.y);
    }
    __syncthreads();

    int lane  = threadIdx.x & 31;
    int wave  = threadIdx.x >> 5;
    int node0 = blockIdx.x * 128 + wave * 16;
    if (node0 >= N) return;                 // wave-uniform: EXEC stays all-1s

    int m  = lane & 15;
    int hi = lane >> 4;

    int nodeA = node0 + m;
    if (nodeA >= N) nodeA = N - 1;          // clamp (N%16==0 so never hit)
    float c   = cnt[nodeA];
    float inv = 1.0f / fmaxf(c, 1.0f);
    const float* aggrow = agg + (long long)nodeA * DFEAT;
    const float* xrow   = xin + (long long)nodeA * DFEAT;

    bf16x16 Am[2], Ax[2];
#pragma unroll
    for (int kh = 0; kh < 2; ++kh) {
        Am[kh] = load_a(aggrow, inv,  kh * 32, hi);
        Ax[kh] = load_a(xrow,   1.0f, kh * 32, hi);
    }

    // one base pointer; j0 + r*DFEAT fold into store instruction offsets
    float* po = out + (long long)(node0 + hi * 8) * DFEAT + m;
    const bool full = (node0 + 16 <= N);    // wave-uniform

#pragma unroll
    for (int j0 = 0; j0 < DFEAT; j0 += 16) {
        f32x8 acc = {};
#pragma unroll
        for (int kh = 0; kh < 2; ++kh) {
            // B[K][n] = w[j0+n][K]; lane covers K = kh*32 + hi*16 .. +15
            int boff = (j0 + m) * DFEAT + kh * 32 + hi * 16;
            bf16x16 Bl = *(const bf16x16*)(swl + boff);
            acc = __builtin_amdgcn_wmma_f32_16x16x32_bf16(
                false, Am[kh], false, Bl, (short)0, acc, false, false);
            bf16x16 Br = *(const bf16x16*)(swr + boff);
            acc = __builtin_amdgcn_wmma_f32_16x16x32_bf16(
                false, Ax[kh], false, Br, (short)0, acc, false, false);
        }
        float bv = bias[j0 + m];
        if (full) {
#pragma unroll
            for (int r = 0; r < 8; ++r) {
                float v = acc[r] + bv;
                if (RELU) v = fmaxf(v, 0.0f);
                po[j0 + r * DFEAT] = v;     // constant offsets after unroll
            }
        } else {
#pragma unroll
            for (int r = 0; r < 8; ++r) {
                float v = acc[r] + bv;
                if (RELU) v = fmaxf(v, 0.0f);
                int row = node0 + hi * 8 + r;
                if (row < N) out[(long long)row * DFEAT + j0 + m] = v;
            }
        }
    }
}

extern "C" void kernel_launch(void* const* d_in, const int* in_sizes, int n_in,
                              void* d_out, int out_size, void* d_ws, size_t ws_size,
                              hipStream_t stream) {
    const float*     x    = (const float*)d_in[0];
    const long long* ei   = (const long long*)d_in[1];   // int64 [2,E]
    const float*     w1_l = (const float*)d_in[2];
    const float*     b1   = (const float*)d_in[3];
    const float*     w1_r = (const float*)d_in[4];
    const float*     w2_l = (const float*)d_in[5];
    const float*     b2   = (const float*)d_in[6];
    const float*     w2_r = (const float*)d_in[7];
    float* out = (float*)d_out;

    const int N = in_sizes[0] / DFEAT;
    const int E = in_sizes[1] / 2;

    // workspace layout: agg [N*64] | cnt [N] | h [N*64]
    float* agg = (float*)d_ws;
    float* cnt = agg + (long long)N * DFEAT;
    float* h   = cnt + N;

    const long long zn = (long long)N * (DFEAT + 1);     // agg+cnt contiguous
    const int zblocks  = 2048;
    const int sblocks  = (int)(((long long)E * 16 + 255) / 256);
    const int lblocks  = (N + 127) / 128;

    // ---- layer 1 ----
    GNNSAGE_zero<<<zblocks, 256, 0, stream>>>(agg, zn);
    GNNSAGE_scatter<<<sblocks, 256, 0, stream>>>(x, ei, agg, cnt, E);
    GNNSAGE_linear<1><<<lblocks, 256, 0, stream>>>(x, agg, cnt, w1_l, w1_r, b1, h, N);

    // ---- layer 2 ----
    GNNSAGE_zero<<<zblocks, 256, 0, stream>>>(agg, zn);
    GNNSAGE_scatter<<<sblocks, 256, 0, stream>>>(h, ei, agg, cnt, E);
    GNNSAGE_linear<0><<<lblocks, 256, 0, stream>>>(h, agg, cnt, w2_l, w2_r, b2, out, N);
}
